// MixtralDecoderLayer_47949014893025
// MI455X (gfx1250) — compile-verified, matrix operand order checked by reading, not run on hardware
//
#include <hip/hip_runtime.h>
#include <cstdint>
#include <cstddef>

// ---------------- problem constants (from reference) ----------------
#define T_TOK 2048
#define H_DIM 2048
#define NH_   16
#define NKV_  4
#define HD_   128
#define I_DIM 4096
#define E_    8
#define EPS_  1e-5f
#define SM_SCALE 0.08838834764831845f   // HD^-0.5

typedef __bf16 bf16;
typedef __attribute__((ext_vector_type(16))) __bf16 v16bf;
typedef __attribute__((ext_vector_type(8)))  float  v8f;

static __device__ inline v16bf pack16(uint4 a, uint4 b) {
  union { uint4 u[2]; v16bf v; } x;
  x.u[0] = a; x.u[1] = b;
  return x.v;
}

static __device__ inline v8f wmma_bf16(v16bf a, v16bf b, v8f c) {
  // D = A(16x32 bf16) * B(32x16 bf16) + C(16x16 f32)
  return __builtin_amdgcn_wmma_f32_16x16x32_bf16(false, a, false, b, (short)0, c,
                                                 false, false);
}

// CDNA5 async global->LDS copy (16B per lane), tracked by ASYNCcnt.
// lds_addr: wave-relative LDS byte address (low 32 bits of generic pointer).
static __device__ inline void async_copy_b128(unsigned lds_addr, const void* gaddr) {
  asm volatile("global_load_async_to_lds_b128 %0, %1, off"
               :: "v"(lds_addr), "v"(gaddr) : "memory");
}
template <int N>
static __device__ inline void wait_async() {
  asm volatile("s_wait_asynccnt %0" :: "i"(N) : "memory");
}
static __device__ inline unsigned lds_addr_of(const void* p) {
  return (unsigned)(size_t)p;   // LDS aperture: addr[31:0] == LDS offset
}

// ---------------------------------------------------------------------------
// Fused residual-add + RMSNorm. One block (256 thr) per token; H=2048.
// ---------------------------------------------------------------------------
__global__ __launch_bounds__(256)
void fused_add_rmsnorm(const float* __restrict__ x, const float* __restrict__ resin,
                       const float* __restrict__ w, float* __restrict__ resout,
                       bf16* __restrict__ hout) {
  const int t = blockIdx.x;
  const int tid = threadIdx.x;
  __shared__ float red[8];
  float v[8];
  float ss = 0.f;
#pragma unroll
  for (int i = 0; i < 8; ++i) {
    int col = tid + i * 256;
    size_t idx = (size_t)t * H_DIM + col;
    float a = x[idx] + resin[idx];
    v[i] = a;
    resout[idx] = a;
    ss += a * a;
  }
#pragma unroll
  for (int m = 16; m >= 1; m >>= 1) ss += __shfl_xor(ss, m);
  if ((tid & 31) == 0) red[tid >> 5] = ss;
  __syncthreads();
  float tot = 0.f;
#pragma unroll
  for (int i = 0; i < 8; ++i) tot += red[i];
  const float inv = rsqrtf(tot / (float)H_DIM + EPS_);
#pragma unroll
  for (int i = 0; i < 8; ++i) {
    int col = tid + i * 256;
    hout[(size_t)t * H_DIM + col] = (bf16)(v[i] * inv * w[col]);
  }
}

// ---------------------------------------------------------------------------
// WMMA GEMM: C[M,N] = A_bf16[M,K] @ B_fp32[N,K]^T  (fp32 accumulate)
// Block tile 128x128, BK=64, 8 waves; wave tile 32x64 = 2x4 WMMA tiles.
// Double-buffered LDS staging:
//   - A tile DMA'd with GLOBAL_LOAD_ASYNC_TO_LDS_B128 (ASYNCcnt pipelined:
//     next tile's 4 DMAs stay in flight while s_wait_asynccnt 4 releases the
//     current tile -> DMA overlaps the 16 WMMAs of the live tile)
//   - B tile streamed fp32 and converted to bf16 while filling LDS.
// EPI: 0 = store fp32, 1 = store bf16, 2 = C += scale[row*sstride] * acc.
// Requires M%128==0, N%128==0, K%64==0 (true for all uses here).
// ---------------------------------------------------------------------------
template <int EPI>
__global__ __launch_bounds__(256)
void wmma_gemm(const bf16* __restrict__ A, const float* __restrict__ B,
               void* __restrict__ Cv, const float* __restrict__ scale,
               int M, int N, int K, int sstride) {
  __shared__ bf16 As[2][128][72];   // 64 + 8 halves pad (row stride 144B)
  __shared__ bf16 Bs[2][128][72];

  const int tid = threadIdx.x;
  const int m0 = blockIdx.y * 128, n0 = blockIdx.x * 128;
  const int wave = tid >> 5, lane = tid & 31;
  const int wm = (wave & 3) * 32, wn = (wave >> 2) * 64;
  const int hi = lane >> 4, lr = lane & 15;
  const int klo = hi * 8;

  // Per-thread A async-copy geometry: 128x64 halves = 1024 x 16B segments,
  // 4 per thread. seg = tid + s*256 -> row = seg>>3, col = (seg&7)*8.
  const int ar = tid >> 3, ac = (tid & 7) * 8;

  auto prefetch = [&](int k0, int buf) {
#pragma unroll
    for (int s = 0; s < 4; ++s)
      async_copy_b128(lds_addr_of(&As[buf][ar + s * 32][ac]),
                      A + (size_t)(m0 + ar + s * 32) * K + k0 + ac);
#pragma unroll
    for (int s = 0; s < 8; ++s) {
      int seg = tid + s * 256;
      int r = seg >> 4, c = (seg & 15) * 4;
      float4 f = *(const float4*)(B + (size_t)(n0 + r) * K + k0 + c);
      Bs[buf][r][c + 0] = (bf16)f.x; Bs[buf][r][c + 1] = (bf16)f.y;
      Bs[buf][r][c + 2] = (bf16)f.z; Bs[buf][r][c + 3] = (bf16)f.w;
    }
  };

  v8f acc[2][4] = {};

  const int nk = K >> 6;
  prefetch(0, 0);
  for (int i = 0; i < nk; ++i) {
    const int cur = i & 1;
    // all waves have finished reading buffer cur^1 (their previous compute)
    __syncthreads();
    if (i + 1 < nk) {
      prefetch((i + 1) << 6, cur ^ 1);
      wait_async<4>();   // in-order ASYNCcnt: <=4 outstanding => cur tile landed
    } else {
      wait_async<0>();
    }
    __syncthreads();     // every wave's cur-tile data visible in LDS

#pragma unroll
    for (int kk = 0; kk < 64; kk += 32) {
      v16bf af[2], bfm[4];
#pragma unroll
      for (int ii = 0; ii < 2; ++ii) {
        int row = wm + ii * 16 + lr;
        af[ii] = pack16(*(const uint4*)(&As[cur][row][kk + klo]),
                        *(const uint4*)(&As[cur][row][kk + klo + 16]));
      }
#pragma unroll
      for (int j = 0; j < 4; ++j) {
        int row = wn + j * 16 + lr;
        bfm[j] = pack16(*(const uint4*)(&Bs[cur][row][kk + klo]),
                        *(const uint4*)(&Bs[cur][row][kk + klo + 16]));
      }
#pragma unroll
      for (int ii = 0; ii < 2; ++ii)
#pragma unroll
        for (int j = 0; j < 4; ++j)
          acc[ii][j] = wmma_bf16(af[ii], bfm[j], acc[ii][j]);
    }
  }

#pragma unroll
  for (int i = 0; i < 2; ++i)
#pragma unroll
    for (int j = 0; j < 4; ++j)
#pragma unroll
      for (int r = 0; r < 8; ++r) {
        int row = m0 + wm + i * 16 + r + 8 * hi;   // C layout: M = r + 8*(lane/16)
        int col = n0 + wn + j * 16 + lr;           //           N = lane%16
        size_t idx = (size_t)row * N + col;
        if (EPI == 0) {
          ((float*)Cv)[idx] = acc[i][j][r];
        } else if (EPI == 1) {
          ((bf16*)Cv)[idx] = (bf16)acc[i][j][r];
        } else {
          float sc = scale[(size_t)row * sstride];
          ((float*)Cv)[idx] += sc * acc[i][j][r];
        }
      }
}

// ---------------------------------------------------------------------------
// NeoX RoPE (fp32 in -> bf16 out). One block per (token, head), 64 threads.
// ---------------------------------------------------------------------------
__global__ __launch_bounds__(64)
void rope_to_bf16(const float* __restrict__ x, const int* __restrict__ pos,
                  bf16* __restrict__ out, int nheads) {
  const int b = blockIdx.x;           // t*nheads + h
  const int t = b / nheads;
  const int d = threadIdx.x;          // 0..63
  const float p = (float)pos[t];
  const float invf = __powf(10000.f, -((float)d) / 64.f);
  float s, c;
  __sincosf(p * invf, &s, &c);
  const size_t base = (size_t)b * HD_;
  const float x1 = x[base + d], x2 = x[base + d + 64];
  out[base + d]      = (bf16)(x1 * c - x2 * s);
  out[base + d + 64] = (bf16)(x2 * c + x1 * s);
}

// V [T, NKV*HD] fp32 -> Vt [NKV*HD, T] bf16 (contiguous in keys for B-operand)
__global__ __launch_bounds__(256)
void transpose_to_bf16(const float* __restrict__ v, bf16* __restrict__ vt) {
  const int id = blockIdx.x * 256 + threadIdx.x;   // over (NKV*HD)*T
  const int d = id / T_TOK, t = id % T_TOK;
  vt[(size_t)d * T_TOK + t] = (bf16)v[(size_t)t * (NKV_ * HD_) + d];
}

// ---------------------------------------------------------------------------
// Flash attention (causal, GQA rep=4). One wave per (16-query tile, head).
// 32-key tiles: S = two 16x16 WMMA accumulators; P@V uses full K=32 WMMA.
// P is bounced through LDS to convert C-layout -> A-operand layout.
// ---------------------------------------------------------------------------
__global__ __launch_bounds__(32)
void flash_attn(const bf16* __restrict__ Q, const bf16* __restrict__ Kc,
                const bf16* __restrict__ Vt, bf16* __restrict__ O) {
  const int qt = blockIdx.x, h = blockIdx.y;
  const int q0 = qt * 16;
  const int kvh = h >> 2;                       // NH/NKV = 4
  const int lane = threadIdx.x;
  const int hi = lane >> 4, lr = lane & 15;
  const int klo = hi * 8;
  __shared__ bf16 Pl[16][40];

  // Q fragments: 4 K-chunks of 32 over HD=128
  v16bf qf[4];
  const bf16* qbase = Q + (size_t)(q0 + lr) * (NH_ * HD_) + h * HD_;
#pragma unroll
  for (int c = 0; c < 4; ++c)
    qf[c] = pack16(*(const uint4*)(qbase + c * 32 + klo),
                   *(const uint4*)(qbase + c * 32 + klo + 16));

  v8f o[8] = {};
  float m[8], l[8];
#pragma unroll
  for (int r = 0; r < 8; ++r) { m[r] = -1e30f; l[r] = 0.f; }

  const int nkt = (qt >> 1) + 1;                // keys 0 .. q0+15 in tiles of 32
  for (int kt = 0; kt < nkt; ++kt) {
    const int kb0 = kt * 32;
    v8f s0 = {}, s1 = {};
    {
      const bf16* kbase = Kc + (size_t)(kb0 + lr) * (NKV_ * HD_) + kvh * HD_;
#pragma unroll
      for (int c = 0; c < 4; ++c)
        s0 = wmma_bf16(qf[c],
                       pack16(*(const uint4*)(kbase + c * 32 + klo),
                              *(const uint4*)(kbase + c * 32 + klo + 16)), s0);
    }
    {
      const bf16* kbase = Kc + (size_t)(kb0 + 16 + lr) * (NKV_ * HD_) + kvh * HD_;
#pragma unroll
      for (int c = 0; c < 4; ++c)
        s1 = wmma_bf16(qf[c],
                       pack16(*(const uint4*)(kbase + c * 32 + klo),
                              *(const uint4*)(kbase + c * 32 + klo + 16)), s1);
    }

    float p0[8], p1[8], alpha[8];
#pragma unroll
    for (int r = 0; r < 8; ++r) {
      const int qrow = q0 + r + 8 * hi;
      float v0 = ((kb0 + lr)      <= qrow) ? s0[r] * SM_SCALE : -1e9f;
      float v1 = ((kb0 + 16 + lr) <= qrow) ? s1[r] * SM_SCALE : -1e9f;
      float rm = fmaxf(v0, v1);
#pragma unroll
      for (int msk = 8; msk >= 1; msk >>= 1) rm = fmaxf(rm, __shfl_xor(rm, msk));
      const float mn = fmaxf(m[r], rm);
      alpha[r] = __expf(m[r] - mn);
      p0[r] = __expf(v0 - mn);
      p1[r] = __expf(v1 - mn);
      float rs = p0[r] + p1[r];
#pragma unroll
      for (int msk = 8; msk >= 1; msk >>= 1) rs += __shfl_xor(rs, msk);
      l[r] = l[r] * alpha[r] + rs;
      m[r] = mn;
    }
#pragma unroll
    for (int j = 0; j < 8; ++j)
#pragma unroll
      for (int r = 0; r < 8; ++r) o[j][r] *= alpha[r];

    // C-layout P -> LDS -> A-operand layout (16x32 bf16)
    __syncthreads();
#pragma unroll
    for (int r = 0; r < 8; ++r) {
      Pl[r + 8 * hi][lr]      = (bf16)p0[r];
      Pl[r + 8 * hi][16 + lr] = (bf16)p1[r];
    }
    __syncthreads();
    v16bf pf = pack16(*(const uint4*)(&Pl[lr][klo]),
                      *(const uint4*)(&Pl[lr][klo + 16]));

    // O += P @ V ; Vt row = d, contiguous keys -> b128 loads
#pragma unroll
    for (int j = 0; j < 8; ++j) {
      const bf16* vbase = Vt + (size_t)(kvh * HD_ + j * 16 + lr) * T_TOK + kb0;
      v16bf vfr = pack16(*(const uint4*)(vbase + klo),
                         *(const uint4*)(vbase + klo + 16));
      o[j] = wmma_bf16(pf, vfr, o[j]);
    }
  }

#pragma unroll
  for (int r = 0; r < 8; ++r) {
    const float inv = 1.f / l[r];
    const int row = q0 + r + 8 * hi;
#pragma unroll
    for (int j = 0; j < 8; ++j)
      O[(size_t)row * (NH_ * HD_) + h * HD_ + j * 16 + lr] = (bf16)(o[j][r] * inv);
  }
}

// ---------------------------------------------------------------------------
// Router: logits [T,E] then per-token softmax + top-2 renormalized coef [T,E]
// ---------------------------------------------------------------------------
__global__ __launch_bounds__(256)
void router_logits(const bf16* __restrict__ h, const float* __restrict__ gw,
                   float* __restrict__ logits) {
  const int id = blockIdx.x * 256 + threadIdx.x;
  if (id >= T_TOK * E_) return;
  const int t = id / E_, e = id % E_;
  const bf16* hp = h + (size_t)t * H_DIM;
  const float* gp = gw + (size_t)e * H_DIM;
  float acc = 0.f;
#pragma unroll 4
  for (int k = 0; k < H_DIM; ++k) acc += (float)hp[k] * gp[k];
  logits[id] = acc;
}

__global__ __launch_bounds__(256)
void router_topk(const float* __restrict__ logits, float* __restrict__ coef) {
  const int t = blockIdx.x * 256 + threadIdx.x;
  if (t >= T_TOK) return;
  float lg[E_], p[E_];
  float mx = -1e30f;
#pragma unroll
  for (int e = 0; e < E_; ++e) { lg[e] = logits[t * E_ + e]; mx = fmaxf(mx, lg[e]); }
  float sum = 0.f;
#pragma unroll
  for (int e = 0; e < E_; ++e) { p[e] = __expf(lg[e] - mx); sum += p[e]; }
#pragma unroll
  for (int e = 0; e < E_; ++e) p[e] /= sum;
  int i1 = 0;
#pragma unroll
  for (int e = 1; e < E_; ++e) if (p[e] > p[i1]) i1 = e;
  int i2 = (i1 == 0) ? 1 : 0;
#pragma unroll
  for (int e = 0; e < E_; ++e) if (e != i1 && p[e] > p[i2]) i2 = e;
  const float wsum = p[i1] + p[i2];
#pragma unroll
  for (int e = 0; e < E_; ++e) coef[t * E_ + e] = 0.f;
  coef[t * E_ + i1] = p[i1] / wsum;
  coef[t * E_ + i2] = p[i2] / wsum;
}

__global__ __launch_bounds__(256)
void silu_mul(const float* __restrict__ g, const float* __restrict__ u,
              bf16* __restrict__ act) {
  const size_t id = (size_t)blockIdx.x * 256 + threadIdx.x;
  const float a = g[id];
  act[id] = (bf16)((a / (1.f + __expf(-a))) * u[id]);
}

// ---------------------------------------------------------------------------
extern "C" void kernel_launch(void* const* d_in, const int* in_sizes, int n_in,
                              void* d_out, int out_size, void* d_ws, size_t ws_size,
                              hipStream_t stream) {
  const int*   positions = (const int*)  d_in[0];
  const float* hidden    = (const float*)d_in[1];
  const float* resid_in  = (const float*)d_in[2];
  const float* ln1_w     = (const float*)d_in[3];
  const float* ln2_w     = (const float*)d_in[4];
  const float* wq        = (const float*)d_in[5];
  const float* wk        = (const float*)d_in[6];
  const float* wv        = (const float*)d_in[7];
  const float* wo        = (const float*)d_in[8];
  const float* gate_w    = (const float*)d_in[9];
  const float* w1        = (const float*)d_in[10];
  const float* w3        = (const float*)d_in[11];
  const float* w2        = (const float*)d_in[12];
  (void)in_sizes; (void)n_in; (void)out_size; (void)ws_size;

  float* out_f   = (float*)d_out;                              // [T,H] MoE output
  float* res_out = (float*)d_out + (size_t)T_TOK * H_DIM;      // [T,H] residual

  char* ws = (char*)d_ws;
  size_t off = 0;
  auto alloc = [&](size_t bytes) -> void* {
    void* p = ws + off;
    off += (bytes + 255) & ~(size_t)255;
    return p;
  };
  float* res1   = (float*)alloc((size_t)T_TOK * H_DIM * 4);
  bf16*  h1     = (bf16*) alloc((size_t)T_TOK * H_DIM * 2);
  float* qf_    = (float*)alloc((size_t)T_TOK * NH_  * HD_ * 4);
  float* kf_    = (float*)alloc((size_t)T_TOK * NKV_ * HD_ * 4);
  float* vf_    = (float*)alloc((size_t)T_TOK * NKV_ * HD_ * 4);
  bf16*  qb     = (bf16*) alloc((size_t)T_TOK * NH_  * HD_ * 2);
  bf16*  kb     = (bf16*) alloc((size_t)T_TOK * NKV_ * HD_ * 2);
  bf16*  vt     = (bf16*) alloc((size_t)NKV_ * HD_ * T_TOK * 2);
  bf16*  attnb  = (bf16*) alloc((size_t)T_TOK * NH_ * HD_ * 2);
  float* of_    = (float*)alloc((size_t)T_TOK * H_DIM * 4);
  bf16*  h2     = (bf16*) alloc((size_t)T_TOK * H_DIM * 2);
  float* logits = (float*)alloc((size_t)T_TOK * E_ * 4);
  float* coef   = (float*)alloc((size_t)T_TOK * E_ * 4);
  float* gbuf   = (float*)alloc((size_t)T_TOK * I_DIM * 4);
  float* ubuf   = (float*)alloc((size_t)T_TOK * I_DIM * 4);
  bf16*  act    = (bf16*) alloc((size_t)T_TOK * I_DIM * 2);

  hipMemsetAsync(out_f, 0, (size_t)T_TOK * H_DIM * sizeof(float), stream);

  // x = hidden + residual; res1 = x; h1 = rmsnorm(x)*ln1_w (bf16)
  fused_add_rmsnorm<<<T_TOK, 256, 0, stream>>>(hidden, resid_in, ln1_w, res1, h1);

  const dim3 blk(256);
  // QKV projections
  wmma_gemm<0><<<dim3((NH_  * HD_) / 128, T_TOK / 128), blk, 0, stream>>>(
      h1, wq, qf_, nullptr, T_TOK, NH_ * HD_, H_DIM, 0);
  wmma_gemm<0><<<dim3((NKV_ * HD_) / 128, T_TOK / 128), blk, 0, stream>>>(
      h1, wk, kf_, nullptr, T_TOK, NKV_ * HD_, H_DIM, 0);
  wmma_gemm<0><<<dim3((NKV_ * HD_) / 128, T_TOK / 128), blk, 0, stream>>>(
      h1, wv, vf_, nullptr, T_TOK, NKV_ * HD_, H_DIM, 0);

  rope_to_bf16<<<T_TOK * NH_,  64, 0, stream>>>(qf_, positions, qb, NH_);
  rope_to_bf16<<<T_TOK * NKV_, 64, 0, stream>>>(kf_, positions, kb, NKV_);
  transpose_to_bf16<<<(NKV_ * HD_ * T_TOK) / 256, 256, 0, stream>>>(vf_, vt);

  flash_attn<<<dim3(T_TOK / 16, NH_), 32, 0, stream>>>(qb, kb, vt, attnb);

  // o_proj
  wmma_gemm<0><<<dim3(H_DIM / 128, T_TOK / 128), blk, 0, stream>>>(
      attnb, wo, of_, nullptr, T_TOK, H_DIM, NH_ * HD_, 0);

  // x = o + res1; residual out goes straight into d_out; h2 = rmsnorm(x)*ln2_w
  fused_add_rmsnorm<<<T_TOK, 256, 0, stream>>>(of_, res1, ln2_w, res_out, h2);

  // router
  router_logits<<<(T_TOK * E_ + 255) / 256, 256, 0, stream>>>(h2, gate_w, logits);
  router_topk<<<(T_TOK + 255) / 256, 256, 0, stream>>>(logits, coef);

  // dense MoE (coef is exactly zero for unselected experts, matching reference)
  for (int e = 0; e < E_; ++e) {
    const float* w1e = w1 + (size_t)e * I_DIM * H_DIM;
    const float* w3e = w3 + (size_t)e * I_DIM * H_DIM;
    const float* w2e = w2 + (size_t)e * H_DIM * I_DIM;
    wmma_gemm<0><<<dim3(I_DIM / 128, T_TOK / 128), blk, 0, stream>>>(
        h2, w1e, gbuf, nullptr, T_TOK, I_DIM, H_DIM, 0);
    wmma_gemm<0><<<dim3(I_DIM / 128, T_TOK / 128), blk, 0, stream>>>(
        h2, w3e, ubuf, nullptr, T_TOK, I_DIM, H_DIM, 0);
    silu_mul<<<(T_TOK * I_DIM) / 256, 256, 0, stream>>>(gbuf, ubuf, act);
    wmma_gemm<2><<<dim3(H_DIM / 128, T_TOK / 128), blk, 0, stream>>>(
        act, w2e, out_f, coef + e, T_TOK, H_DIM, I_DIM, E_);
  }
}
